// RNNEncoder_24129126269381
// MI455X (gfx1250) — matrix operand order repaired and checked by reading, not compiled, not run on hardware
//
#include <hip/hip_runtime.h>

// ---------------------------------------------------------------------------
// Fused 2-layer GRU (H=64) over 512 independent sequences, L=2048.
// One workgroup = 16 sequences (one WMMA M-tile), 4 waves; wave w owns gate
// columns [48w, 48w+48) (3 WMMA N-tiles). Weights resident in VGPRs as WMMA
// B-fragments; hidden state resident in LDS; per-step 18 v_wmma per wave.
// Gate nonlinearities use the TRANS unit (v_exp_f32/v_rcp_f32) instead of
// precise division / ocml tanh to shorten the serial per-step chain.
// ---------------------------------------------------------------------------

typedef __attribute__((ext_vector_type(16))) _Float16 v16h;
typedef __attribute__((ext_vector_type(8)))  float    v8f;
typedef __attribute__((ext_vector_type(4)))  float    v4f;   // for NT stores

#define H        64
#define G        192            // 3*H gate width
#define LSEQ     2048
#define CCH      32             // channels in x
#define TILE_SEQ 16
#define THREADS  128
#define HGS      196            // padded f32 stride for gate buffers (bank-safe)

#define WMMA_F16(A, B, C) \
    __builtin_amdgcn_wmma_f32_16x16x32_f16(false, (A), false, (B), (short)0, (C), false, false)

// v_exp_f32 + v_rcp_f32: 2 TRANS ops, co-executes with VALU.
__device__ __forceinline__ float sigmoid_f(float x) {
    return __builtin_amdgcn_rcpf(1.0f + __expf(-x));
}

// tanh(x) = 1 - 2/(exp(2x)+1); saturates to +/-1 when exp over/underflows.
__device__ __forceinline__ float tanh_f(float x) {
    return 1.0f - 2.0f * __builtin_amdgcn_rcpf(__expf(2.0f * x) + 1.0f);
}

// B-fragment for D = A(16xK) * B(Kx16): lane&15 = output column; low lanes hold
// K k0..k0+7 and k0+16..k0+23, high lanes the +8 offsets. hg = h @ W^T means
// B[k][col] = W[col][k], so each lane reads two 8-float runs of one weight row.
__device__ __forceinline__ v16h load_bfrag(const float* __restrict__ W,
                                           int ncol_base, int kblock, int lane) {
    const int col = ncol_base + (lane & 15);
    const int k0  = kblock * 32 + ((lane & 16) ? 8 : 0);
    const float* p = W + col * H + k0;
    v16h f;
#pragma unroll
    for (int i = 0; i < 16; ++i) f[i] = (_Float16)p[i < 8 ? i : i + 8];
    return f;
}

// A-fragment (16x32 f16): lane&15 = row (sequence); low lanes hold K k0..k0+7
// and k0+16..k0+23, high lanes the +8 offsets — two ds_load_b128 each.
__device__ __forceinline__ v16h load_afrag(const _Float16* sh, int kblock, int lane) {
    const int row = lane & 15;
    const int k0  = kblock * 32 + ((lane & 16) ? 8 : 0);
    const _Float16* p = sh + row * H + k0;
    v16h f;
    ((uint4*)&f)[0] = *(const uint4*)(p);
    ((uint4*)&f)[1] = *(const uint4*)(p + 16);
    return f;
}

// C/D fragment (16x16 f32, 8 VGPRs): lane&15 = N column, VGPR i = row (+8 for
// high lanes). Scatter into the padded gate buffer.
__device__ __forceinline__ void store_acc(float* dst, v8f acc, int ncol_base, int lane) {
    const int col  = ncol_base + (lane & 15);
    const int mrow = (lane & 16) ? 8 : 0;
#pragma unroll
    for (int i = 0; i < 8; ++i) dst[(mrow + i) * HGS + col] = acc[i];
}

__global__ __launch_bounds__(THREADS, 1)
void gru2_fused_kernel(const float* __restrict__ x,
                       const float* __restrict__ w_ih0,
                       const float* __restrict__ w_hh0,
                       const float* __restrict__ b0,
                       const float* __restrict__ bn0,
                       const float* __restrict__ w_ih1,
                       const float* __restrict__ w_hh1,
                       const float* __restrict__ b1,
                       const float* __restrict__ bn1,
                       float* __restrict__ out) {
    __shared__ __align__(16) _Float16 sh_h0[TILE_SEQ * H];   // f16 h0 (WMMA A src)
    __shared__ __align__(16) _Float16 sh_h1[TILE_SEQ * H];   // f16 h1 (WMMA A src)
    __shared__ __align__(16) float sh_h0f[TILE_SEQ * H];     // f32 master h0
    __shared__ __align__(16) float sh_h1f[TILE_SEQ * H];     // f32 master h1
    __shared__ __align__(16) float sh_hg[TILE_SEQ * HGS];    // hg0 then hg1
    __shared__ __align__(16) float sh_ig[TILE_SEQ * HGS];    // ig1
    __shared__ float sh_xv[TILE_SEQ];
    __shared__ float sh_wih0[G], sh_b0[G], sh_b1[G], sh_bn0[H], sh_bn1[H];

    const int tid  = threadIdx.x;
    const int lane = tid & 31;
    const int wv   = tid >> 5;          // wave 0..3
    const int wg   = blockIdx.x;        // 0..31 sequence tiles

    // ---- one-time staging: small params + zero hidden state ----------------
    for (int i = tid; i < G; i += THREADS) {
        sh_wih0[i] = w_ih0[i];
        sh_b0[i]   = b0[i];
        sh_b1[i]   = b1[i];
    }
    for (int i = tid; i < H; i += THREADS) {
        sh_bn0[i] = bn0[i];
        sh_bn1[i] = bn1[i];
    }
    for (int i = tid; i < TILE_SEQ * H; i += THREADS) {
        sh_h0[i]  = (_Float16)0.0f;
        sh_h1[i]  = (_Float16)0.0f;
        sh_h0f[i] = 0.0f;
        sh_h1f[i] = 0.0f;
    }

    // ---- register-resident weight B-fragments (loaded once, used 2048x) ----
    v16h whh0[3][2], wih1[3][2], whh1[3][2];
#pragma unroll
    for (int i = 0; i < 3; ++i) {
        const int nb = (3 * wv + i) * 16;
#pragma unroll
        for (int kb = 0; kb < 2; ++kb) {
            whh0[i][kb] = load_bfrag(w_hh0, nb, kb, lane);
            wih1[i][kb] = load_bfrag(w_ih1, nb, kb, lane);
            whh1[i][kb] = load_bfrag(w_hh1, nb, kb, lane);
        }
    }
    __syncthreads();

    // x is (B=16, L, C=32); sequence n = b*32 + c. This block owns 16
    // consecutive c's of one b, so x_t for the tile is 16 contiguous floats.
    const int bidx = wg >> 1;
    const int c0   = (wg & 1) * 16;
    const float* xblk = x + ((size_t)bidx * LSEQ) * CCH + c0;

    // gate-phase mapping: thread -> (sequence, 8 contiguous hidden units)
    const int s_my  = tid >> 3;
    const int jb_my = (tid & 7) * 8;
    const size_t orow_base =
        (((size_t)wg * TILE_SEQ + (size_t)s_my) * LSEQ) * H + (size_t)jb_my;

    for (int t = 0; t < LSEQ; ++t) {
        // ---- phase A: hg0 = h0 @ Whh0^T (WMMA), stage x_t -----------------
        {
            v16h a0 = load_afrag(sh_h0, 0, lane);
            v16h a1 = load_afrag(sh_h0, 1, lane);
#pragma unroll
            for (int i = 0; i < 3; ++i) {
                v8f acc = {};
                acc = WMMA_F16(a0, whh0[i][0], acc);
                acc = WMMA_F16(a1, whh0[i][1], acc);
                store_acc(sh_hg, acc, (3 * wv + i) * 16, lane);
            }
        }
        if (tid < TILE_SEQ) {
            sh_xv[tid] = xblk[(size_t)t * CCH + tid];
            if (t + 1 < LSEQ)
                __builtin_prefetch(&xblk[(size_t)(t + 1) * CCH + tid], 0, 1);
        }
        __syncthreads();

        // ---- phase B: layer-0 gate math, update h0 ------------------------
        {
            const float xv = sh_xv[s_my];
#pragma unroll
            for (int u = 0; u < 8; ++u) {
                const int j = jb_my + u;
                const float igr = fmaf(xv, sh_wih0[j],         sh_b0[j]);
                const float igz = fmaf(xv, sh_wih0[H + j],     sh_b0[H + j]);
                const float ign = fmaf(xv, sh_wih0[2 * H + j], sh_b0[2 * H + j]);
                const float hgr = sh_hg[s_my * HGS + j];
                const float hgz = sh_hg[s_my * HGS + H + j];
                const float hgn = sh_hg[s_my * HGS + 2 * H + j];
                const float r = sigmoid_f(igr + hgr);
                const float z = sigmoid_f(igz + hgz);
                const float n = tanh_f(ign + r * (hgn + sh_bn0[j]));
                const float h  = sh_h0f[s_my * H + j];
                const float hn = n + z * (h - n);
                sh_h0f[s_my * H + j] = hn;
                sh_h0[s_my * H + j]  = (_Float16)hn;
            }
        }
        __syncthreads();

        // ---- phase C: ig1 = h0' @ Wih1^T ; hg1 = h1 @ Whh1^T --------------
        {
            v16h p0 = load_afrag(sh_h0, 0, lane);
            v16h p1 = load_afrag(sh_h0, 1, lane);
            v16h q0 = load_afrag(sh_h1, 0, lane);
            v16h q1 = load_afrag(sh_h1, 1, lane);
#pragma unroll
            for (int i = 0; i < 3; ++i) {
                const int nb = (3 * wv + i) * 16;
                v8f accI = {};
                accI = WMMA_F16(p0, wih1[i][0], accI);
                accI = WMMA_F16(p1, wih1[i][1], accI);
                store_acc(sh_ig, accI, nb, lane);
                v8f accH = {};
                accH = WMMA_F16(q0, whh1[i][0], accH);
                accH = WMMA_F16(q1, whh1[i][1], accH);
                store_acc(sh_hg, accH, nb, lane);
            }
        }
        __syncthreads();

        // ---- phase D: layer-1 gates, update h1, stream output -------------
        {
            float hout[8];
#pragma unroll
            for (int u = 0; u < 8; ++u) {
                const int j = jb_my + u;
                const float igr = sh_ig[s_my * HGS + j]         + sh_b1[j];
                const float igz = sh_ig[s_my * HGS + H + j]     + sh_b1[H + j];
                const float ign = sh_ig[s_my * HGS + 2 * H + j] + sh_b1[2 * H + j];
                const float hgr = sh_hg[s_my * HGS + j];
                const float hgz = sh_hg[s_my * HGS + H + j];
                const float hgn = sh_hg[s_my * HGS + 2 * H + j];
                const float r = sigmoid_f(igr + hgr);
                const float z = sigmoid_f(igz + hgz);
                const float n = tanh_f(ign + r * (hgn + sh_bn1[j]));
                const float h  = sh_h1f[s_my * H + j];
                const float hn = n + z * (h - n);
                sh_h1f[s_my * H + j] = hn;
                sh_h1[s_my * H + j]  = (_Float16)hn;
                hout[u] = hn;
            }
            // out[b][c][t][:]: 268 MB pure stream -> non-temporal b128 stores
            float* orow = out + orow_base + (size_t)t * H;
            v4f lo = { hout[0], hout[1], hout[2], hout[3] };
            v4f hi = { hout[4], hout[5], hout[6], hout[7] };
            __builtin_nontemporal_store(lo, (v4f*)(orow));
            __builtin_nontemporal_store(hi, (v4f*)(orow + 4));
        }
        __syncthreads();   // protects sh_hg / sh_h1 for the next iteration
    }
}

extern "C" void kernel_launch(void* const* d_in, const int* in_sizes, int n_in,
                              void* d_out, int out_size, void* d_ws, size_t ws_size,
                              hipStream_t stream) {
    const float* x     = (const float*)d_in[0];
    const float* w_ih0 = (const float*)d_in[1];   // (192,1)
    const float* w_hh0 = (const float*)d_in[2];   // (192,64)
    const float* b0    = (const float*)d_in[3];   // (192,)
    const float* bn0   = (const float*)d_in[4];   // (64,)
    const float* w_ih1 = (const float*)d_in[5];   // (192,64)
    const float* w_hh1 = (const float*)d_in[6];   // (192,64)
    const float* b1    = (const float*)d_in[7];   // (192,)
    const float* bn1   = (const float*)d_in[8];   // (64,)
    float* out = (float*)d_out;                   // (16,32,2048,64) f32

    (void)in_sizes; (void)n_in; (void)out_size; (void)d_ws; (void)ws_size;

    dim3 grid(32);          // 512 sequences / 16 per tile
    dim3 block(THREADS);    // 4 waves of 32
    hipLaunchKernelGGL(gru2_fused_kernel, grid, block, 0, stream,
                       x, w_ih0, w_hh0, b0, bn0, w_ih1, w_hh1, b1, bn1, out);
}
